// MoELayer_71605694758956
// MI455X (gfx1250) — compile-verified, matrix operand order checked by reading, not run on hardware
//
#include <hip/hip_runtime.h>
#include <hip/hip_bf16.h>
#include <math.h>

// ---------------- problem constants ----------------
#define HID   1024          // hidden
#define FF    4096          // expert ffn dim
#define NE    8             // experts
#define NTOK  4096          // 2*2048 tokens
#define NSLOT (NTOK * 2)    // total routed (token, expert) slots = T * TOP_K
#define WELEM ((size_t)NE * FF * HID)   // elements per weight tensor

// ---------------- vector types ----------------
typedef __attribute__((ext_vector_type(16))) __bf16 v16bf;
typedef __attribute__((ext_vector_type(8)))  __bf16 v8bf;
typedef __attribute__((ext_vector_type(8)))  float  v8f;

// ---------------- workspace layout (bytes) ----------------
#define WS_CNT      0
#define WS_BASE     64
#define WS_EPOS     128
#define WS_WK       (WS_EPOS + NSLOT * 4)
#define WS_TOK      (WS_WK + NSLOT * 4)
#define WS_XB       ((size_t)((WS_TOK + NE * NTOK * 4 + 255) & ~255))
#define WS_W1B      (WS_XB  + (size_t)NTOK * HID * 2)
#define WS_W3B      (WS_W1B + WELEM * 2)
#define WS_W2B      (WS_W3B + WELEM * 2)
#define WS_H        (WS_W2B + WELEM * 2)
#define WS_CONTRIB  (WS_H + (size_t)NSLOT * FF * 2)
// total ~296 MiB

// ---------------- tiny setup kernels ----------------
__global__ void reset_kernel(int* __restrict__ cnt) {
    if (threadIdx.x < NE) cnt[threadIdx.x] = 0;
}

__global__ void prefix_kernel(const int* __restrict__ cnt, int* __restrict__ basep) {
    if (threadIdx.x == 0) {
        int s = 0;
        for (int e = 0; e < NE; ++e) { basep[e] = s; s += cnt[e]; }
    }
}

// generic f32 -> bf16 streaming convert, 8 elems/thread
__global__ __launch_bounds__(256) void cvt_kernel(const float* __restrict__ src,
                                                  __bf16* __restrict__ dst) {
    size_t i = ((size_t)blockIdx.x * 256 + threadIdx.x) * 8;
    v8f v = *(const v8f*)(src + i);
    v8bf o;
#pragma unroll
    for (int k = 0; k < 8; ++k) o[k] = (__bf16)v[k];
    *(v8bf*)(dst + i) = o;
}

// ---------------- gating: logits -> softmax -> top2 -> bucket append ----------------
__global__ __launch_bounds__(256) void gate_kernel(const float* __restrict__ x,
                                                   const float* __restrict__ gw,
                                                   int* __restrict__ cnt,
                                                   int* __restrict__ tok,
                                                   int* __restrict__ epos,
                                                   float* __restrict__ wk) {
    int t = blockIdx.x * 256 + threadIdx.x;
    if (t >= NTOK) return;

    float acc[NE];
#pragma unroll
    for (int e = 0; e < NE; ++e) acc[e] = 0.f;

    const float* xr = x + (size_t)t * HID;
    for (int i = 0; i < HID; i += 4) {
        float4 xv = *(const float4*)(xr + i);
#pragma unroll
        for (int e = 0; e < NE; ++e) {
            float4 gv = *(const float4*)(gw + (size_t)e * HID + i);
            acc[e] = fmaf(xv.x, gv.x, fmaf(xv.y, gv.y,
                     fmaf(xv.z, gv.z, fmaf(xv.w, gv.w, acc[e]))));
        }
    }
    float mx = acc[0];
#pragma unroll
    for (int e = 1; e < NE; ++e) mx = fmaxf(mx, acc[e]);
#pragma unroll
    for (int e = 0; e < NE; ++e) acc[e] = __expf(acc[e] - mx);

    int i0 = 0; float p0 = acc[0];
#pragma unroll
    for (int e = 1; e < NE; ++e) if (acc[e] > p0) { p0 = acc[e]; i0 = e; }
    int i1 = -1; float p1 = -1.f;
#pragma unroll
    for (int e = 0; e < NE; ++e) if (e != i0 && acc[e] > p1) { p1 = acc[e]; i1 = e; }

    float inv = 1.f / (p0 + p1);       // top-k renorm (softmax denom cancels)
    int pos0 = atomicAdd(cnt + i0, 1);
    tok[i0 * NTOK + pos0] = t;
    epos[2 * t + 0] = (i0 << 16) | pos0;
    wk[2 * t + 0] = p0 * inv;
    int pos1 = atomicAdd(cnt + i1, 1);
    tok[i1 * NTOK + pos1] = t;
    epos[2 * t + 1] = (i1 << 16) | pos1;
    wk[2 * t + 1] = p1 * inv;
}

// ---------------- pipelined fragment helpers ----------------
__device__ __forceinline__ void load_frags1(v16bf a[4], v16bf f1[2], v16bf f3[2],
                                            const __bf16* const arow[4],
                                            const __bf16* const b1r[2],
                                            const __bf16* const b3r[2], int kb) {
#pragma unroll
    for (int i = 0; i < 4; ++i) a[i] = *(const v16bf*)(arow[i] + kb);
#pragma unroll
    for (int j = 0; j < 2; ++j) {
        f1[j] = *(const v16bf*)(b1r[j] + kb);
        f3[j] = *(const v16bf*)(b3r[j] + kb);
    }
}

__device__ __forceinline__ void mma1(v8f c1[4][2], v8f c3[4][2],
                                     const v16bf a[4], const v16bf f1[2],
                                     const v16bf f3[2]) {
#pragma unroll
    for (int i = 0; i < 4; ++i)
#pragma unroll
        for (int j = 0; j < 2; ++j) {
            c1[i][j] = __builtin_amdgcn_wmma_f32_16x16x32_bf16(
                false, a[i], false, f1[j], (short)0, c1[i][j], false, false);
            c3[i][j] = __builtin_amdgcn_wmma_f32_16x16x32_bf16(
                false, a[i], false, f3[j], (short)0, c3[i][j], false, false);
        }
}

__device__ __forceinline__ void load_frags2(v16bf a[4], v16bf b[2],
                                            const __bf16* const arow[4],
                                            const __bf16* const br[2], int kb) {
#pragma unroll
    for (int i = 0; i < 4; ++i) a[i] = *(const v16bf*)(arow[i] + kb);
#pragma unroll
    for (int j = 0; j < 2; ++j) b[j] = *(const v16bf*)(br[j] + kb);
}

__device__ __forceinline__ void mma2(v8f acc[4][2], const v16bf a[4], const v16bf b[2]) {
#pragma unroll
    for (int i = 0; i < 4; ++i)
#pragma unroll
        for (int j = 0; j < 2; ++j)
            acc[i][j] = __builtin_amdgcn_wmma_f32_16x16x32_bf16(
                false, a[i], false, b[j], (short)0, acc[i][j], false, false);
}

// ---------------- GEMM1: h = silu(x @ w1^T) * (x @ w3^T), routed rows ----------------
// block = 128 threads (4 waves). Block tile: 64 token-rows x 128 FF-cols.
// Each wave: 64 rows x 32 cols -> 16 WMMA per K-step vs 16 b128 loads.
// K loop is 2-stage software pipelined to overlap loads with WMMA.
__global__ __launch_bounds__(128) void gemm1_kernel(const __bf16* __restrict__ w1b,
                                                    const __bf16* __restrict__ w3b,
                                                    const __bf16* __restrict__ xb,
                                                    const int* __restrict__ cnt,
                                                    const int* __restrict__ basep,
                                                    const int* __restrict__ tok,
                                                    __bf16* __restrict__ hbuf) {
    const int e = blockIdx.z;
    const int c = cnt[e];
    const int m_tile = blockIdx.y * 64;
    if (m_tile >= c) return;                 // uniform early exit

    const int lane  = threadIdx.x & 31;
    const int wave  = threadIdx.x >> 5;
    const int lhalf = lane >> 4;             // K-half select (WMMA bf16 lane layout)
    const int lrow  = lane & 15;
    const int n_base = blockIdx.x * 128 + wave * 32;

    const __bf16* arow[4];
#pragma unroll
    for (int i = 0; i < 4; ++i) {
        int pos  = m_tile + 16 * i + lrow;
        int posc = pos < c ? pos : c - 1;    // clamp: EXEC all-1s for WMMA
        int t    = tok[e * NTOK + posc];
        arow[i]  = xb + (size_t)t * HID + lhalf * 16;
    }
    const __bf16 *b1r[2], *b3r[2];
#pragma unroll
    for (int j = 0; j < 2; ++j) {
        size_t n = (size_t)(n_base + 16 * j + lrow);
        b1r[j] = w1b + (size_t)e * FF * HID + n * HID + lhalf * 16;
        b3r[j] = w3b + (size_t)e * FF * HID + n * HID + lhalf * 16;
    }

    v8f c1[4][2], c3[4][2];
    v8f zero = {0.f, 0.f, 0.f, 0.f, 0.f, 0.f, 0.f, 0.f};
#pragma unroll
    for (int i = 0; i < 4; ++i)
#pragma unroll
        for (int j = 0; j < 2; ++j) { c1[i][j] = zero; c3[i][j] = zero; }

    v16bf aA[4], f1A[2], f3A[2];
    v16bf aB[4], f1B[2], f3B[2];
    load_frags1(aA, f1A, f3A, arow, b1r, b3r, 0);
    for (int kb = 0; kb < HID - 64; kb += 64) {
        load_frags1(aB, f1B, f3B, arow, b1r, b3r, kb + 32);
        mma1(c1, c3, aA, f1A, f3A);
        load_frags1(aA, f1A, f3A, arow, b1r, b3r, kb + 64);
        mma1(c1, c3, aB, f1B, f3B);
    }
    load_frags1(aB, f1B, f3B, arow, b1r, b3r, HID - 32);
    mma1(c1, c3, aA, f1A, f3A);
    mma1(c1, c3, aB, f1B, f3B);

    const int be = basep[e];
#pragma unroll
    for (int i = 0; i < 4; ++i)
#pragma unroll
        for (int j = 0; j < 2; ++j)
#pragma unroll
            for (int r = 0; r < 8; ++r) {
                float v1 = c1[i][j][r];
                float hv = (v1 / (1.f + __expf(-v1))) * c3[i][j][r];   // SwiGLU
                int p = m_tile + 16 * i + r + 8 * lhalf;  // D layout: M=r+8*(lane>=16)
                if (p < c) {
                    size_t row = (size_t)(be + p);
                    hbuf[row * FF + n_base + 16 * j + lrow] = (__bf16)hv;
                }
            }
}

// ---------------- GEMM2: contrib = h @ w2^T (per routed slot, unweighted) ------------
__global__ __launch_bounds__(128) void gemm2_kernel(const __bf16* __restrict__ w2b,
                                                    const __bf16* __restrict__ hbuf,
                                                    const int* __restrict__ cnt,
                                                    const int* __restrict__ basep,
                                                    float* __restrict__ contrib) {
    const int e = blockIdx.z;
    const int c = cnt[e];
    const int m_tile = blockIdx.y * 64;
    if (m_tile >= c) return;

    const int lane  = threadIdx.x & 31;
    const int wave  = threadIdx.x >> 5;
    const int lhalf = lane >> 4;
    const int lrow  = lane & 15;
    const int n_base = blockIdx.x * 128 + wave * 32;   // within HID
    const int be    = basep[e];

    const __bf16* arow[4];
#pragma unroll
    for (int i = 0; i < 4; ++i) {
        int pos  = m_tile + 16 * i + lrow;
        int posc = pos < c ? pos : c - 1;
        arow[i]  = hbuf + (size_t)(be + posc) * FF + lhalf * 16;
    }
    const __bf16* br[2];
#pragma unroll
    for (int j = 0; j < 2; ++j) {
        size_t n = (size_t)(n_base + 16 * j + lrow);
        br[j] = w2b + (size_t)e * HID * FF + n * FF + lhalf * 16;
    }

    v8f acc[4][2];
    v8f zero = {0.f, 0.f, 0.f, 0.f, 0.f, 0.f, 0.f, 0.f};
#pragma unroll
    for (int i = 0; i < 4; ++i)
#pragma unroll
        for (int j = 0; j < 2; ++j) acc[i][j] = zero;

    v16bf aA[4], bA[2], aB[4], bB[2];
    load_frags2(aA, bA, arow, br, 0);
    for (int kb = 0; kb < FF - 64; kb += 64) {
        load_frags2(aB, bB, arow, br, kb + 32);
        mma2(acc, aA, bA);
        load_frags2(aA, bA, arow, br, kb + 64);
        mma2(acc, aB, bB);
    }
    load_frags2(aB, bB, arow, br, FF - 32);
    mma2(acc, aA, bA);
    mma2(acc, aB, bB);

#pragma unroll
    for (int i = 0; i < 4; ++i)
#pragma unroll
        for (int j = 0; j < 2; ++j)
#pragma unroll
            for (int r = 0; r < 8; ++r) {
                int p = m_tile + 16 * i + r + 8 * lhalf;
                if (p < c) {
                    size_t row = (size_t)(be + p);
                    contrib[row * HID + n_base + 16 * j + lrow] = acc[i][j][r];
                }
            }
}

// ---------------- combine: out[t] = w0*contrib[slot0] + w1*contrib[slot1] ------------
__global__ __launch_bounds__(256) void combine_kernel(const float* __restrict__ contrib,
                                                      const int* __restrict__ epos,
                                                      const float* __restrict__ wk,
                                                      const int* __restrict__ basep,
                                                      float* __restrict__ out) {
    int idx = blockIdx.x * 256 + threadIdx.x;     // over NTOK*HID
    int t = idx >> 10;                            // HID == 1024
    int h = idx & (HID - 1);
    int ep0 = epos[2 * t + 0], ep1 = epos[2 * t + 1];
    float w0 = wk[2 * t + 0],  w1 = wk[2 * t + 1];
    size_t s0 = (size_t)(basep[ep0 >> 16] + (ep0 & 0xFFFF));
    size_t s1 = (size_t)(basep[ep1 >> 16] + (ep1 & 0xFFFF));
    out[idx] = w0 * contrib[s0 * HID + h] + w1 * contrib[s1 * HID + h];
}

// ---------------- launcher ----------------
extern "C" void kernel_launch(void* const* d_in, const int* in_sizes, int n_in,
                              void* d_out, int out_size, void* d_ws, size_t ws_size,
                              hipStream_t stream) {
    (void)in_sizes; (void)n_in; (void)out_size; (void)ws_size;
    const float* x   = (const float*)d_in[0];
    const float* gw  = (const float*)d_in[1];
    const float* w1  = (const float*)d_in[2];
    const float* w2  = (const float*)d_in[3];
    const float* w3  = (const float*)d_in[4];
    float* out = (float*)d_out;

    char* ws = (char*)d_ws;
    int*    cnt     = (int*)(ws + WS_CNT);
    int*    basep   = (int*)(ws + WS_BASE);
    int*    epos    = (int*)(ws + WS_EPOS);
    float*  wkv     = (float*)(ws + WS_WK);
    int*    tok     = (int*)(ws + WS_TOK);
    __bf16* xb      = (__bf16*)(ws + WS_XB);
    __bf16* w1b     = (__bf16*)(ws + WS_W1B);
    __bf16* w3b     = (__bf16*)(ws + WS_W3B);
    __bf16* w2b     = (__bf16*)(ws + WS_W2B);
    __bf16* hbuf    = (__bf16*)(ws + WS_H);
    float*  contrib = (float*)(ws + WS_CONTRIB);

    reset_kernel<<<1, 32, 0, stream>>>(cnt);
    gate_kernel<<<NTOK / 256, 256, 0, stream>>>(x, gw, cnt, tok, epos, wkv);
    prefix_kernel<<<1, 32, 0, stream>>>(cnt, basep);

    cvt_kernel<<<(NTOK * HID) / 2048, 256, 0, stream>>>(x, xb);
    cvt_kernel<<<(int)(WELEM / 2048), 256, 0, stream>>>(w1, w1b);
    cvt_kernel<<<(int)(WELEM / 2048), 256, 0, stream>>>(w3, w3b);
    cvt_kernel<<<(int)(WELEM / 2048), 256, 0, stream>>>(w2, w2b);

    gemm1_kernel<<<dim3(FF / 128, NTOK / 64, NE), 128, 0, stream>>>(
        w1b, w3b, xb, cnt, basep, tok, hbuf);
    gemm2_kernel<<<dim3(HID / 128, NTOK / 64, NE), 128, 0, stream>>>(
        w2b, hbuf, cnt, basep, contrib);
    combine_kernel<<<(NTOK * HID) / 256, 256, 0, stream>>>(
        contrib, epos, wkv, basep, out);
}
